// RNN_54073638256728
// MI455X (gfx1250) — matrix operand order repaired and checked by reading, not compiled, not run on hardware
//
#include <hip/hip_runtime.h>
#include <hip/hip_bf16.h>
#include <math.h>

typedef __bf16 bf16_t;
typedef __attribute__((ext_vector_type(16))) __bf16 v16bf;
typedef __attribute__((ext_vector_type(8)))  __bf16 v8bf;
typedef __attribute__((ext_vector_type(8)))  float  v8f;
typedef __attribute__((ext_vector_type(4)))  int    v4i;

#define T_STEPS 100
#define BATCH   200
#define NGG     4096
#define NPP     512
#define IMGD    512

// CDNA5 async global->LDS path (guarded; falls back to reg-staged copy)
#if defined(__gfx1250__) && __has_builtin(__builtin_amdgcn_global_load_async_to_lds_b128) && \
    __has_builtin(__builtin_amdgcn_s_wait_asynccnt)
#define USE_ASYNC_LDS 1
#else
#define USE_ASYNC_LDS 0
#endif

#define GAS __attribute__((address_space(1)))
#define LAS __attribute__((address_space(3)))

// ------------------------- conversion / packing kernels -------------------------

__global__ void cvt_f32_bf16(const float* __restrict__ src, bf16_t* __restrict__ dst, int n) {
    int i = blockIdx.x * blockDim.x + threadIdx.x;
    if (i < n) dst[i] = (bf16_t)src[i];
}

__global__ void pack_enc_kernel(const float* __restrict__ Wpc, const float* __restrict__ Wimg,
                                bf16_t* __restrict__ dst) {
    int i = blockIdx.x * blockDim.x + threadIdx.x;
    const int n = NGG * 1024;
    if (i < n) {
        int row = i >> 10, col = i & 1023;
        float v = (col < NPP) ? Wpc[row * NPP + col] : Wimg[row * IMGD + (col - NPP)];
        dst[i] = (bf16_t)v;
    }
}

// A0 = [init_actv | image[0]] : [200,1024] bf16 (rows 200..255 left as-is; never stored)
__global__ void pack_a0_kernel(const float* __restrict__ init_actv, const float* __restrict__ image,
                               bf16_t* __restrict__ dst) {
    int i = blockIdx.x * blockDim.x + threadIdx.x;
    const int n = BATCH * 1024;
    if (i < n) {
        int b = i >> 10, k = i & 1023;
        float v = (k < NPP) ? init_actv[b * NPP + k] : image[b * IMGD + (k - NPP)];
        dst[i] = (bf16_t)v;
    }
}

// ------------------------- WMMA GEMM -------------------------
// C[M,N] = A[M,K](bf16,row-major) x Bm[N,K](bf16,row-major == B^T, torch layout)
// Block tile 64(M) x 128(N); 8 waves = 2(M) x 4(N), each wave 32x32 (2x2 WMMA tiles).
// A-side buffers are allocated with >=256 rows, so staging loads are unguarded;
// the epilogue guards stores with gm < M.
// MODE 0: store bf16 (no activation) -> hout
// MODE 1: relu(acc + vel_t . W_v2h)  -> bf16 hout
// MODE 2: decode: n<512 -> place preds; n>=512 -> sigmoid -> image preds

template <int MODE>
__global__ __launch_bounds__(256)
void gemm_wmma(const bf16_t* __restrict__ A, const bf16_t* __restrict__ Bm,
               int M, int K, int t,
               const float* __restrict__ vel, const float* __restrict__ wv2h,
               bf16_t* __restrict__ hout,
               float* __restrict__ out_pl, float* __restrict__ out_im) {
    constexpr int KC  = 32;
    constexpr int LDW = 40;   // padded LDS row stride in halves (80B, 16B aligned)

    const int tid  = threadIdx.x;
    const int lane = tid & 31;
    const int wave = tid >> 5;
    const int wm   = wave & 1;
    const int wn   = wave >> 1;
    const int bm   = blockIdx.y * 64;
    const int bn   = blockIdx.x * 128;

    // staging map: A 64 rows x 4 16B-segs (1/thread), B 128 rows x 4 segs (2/thread)
    const int aRow  = tid >> 2;
    const int seg   = (tid & 3) * 8;      // halves
    const int bRow0 = tid >> 2;
    const int bRow1 = 64 + (tid >> 2);

    const int nk = K / KC;
    const int hf   = lane >> 4;
    const int lrow = lane & 15;

    v8f acc[2][2];
#pragma unroll
    for (int i = 0; i < 2; ++i)
#pragma unroll
        for (int j = 0; j < 2; ++j)
            acc[i][j] = (v8f){0.f, 0.f, 0.f, 0.f, 0.f, 0.f, 0.f, 0.f};

    // fragment build + 4 WMMAs from one staged chunk
    auto compute_chunk = [&](const bf16_t* Ab, const bf16_t* Bb) {
        v16bf afr[2], bfr[2];
#pragma unroll
        for (int i = 0; i < 2; ++i) {
            const bf16_t* p = Ab + (wm * 32 + i * 16 + lrow) * LDW + hf * 8;
            v8bf lo = *(const v8bf*)p;
            v8bf hi = *(const v8bf*)(p + 16);
            afr[i] = __builtin_shufflevector(lo, hi, 0,1,2,3,4,5,6,7,8,9,10,11,12,13,14,15);
        }
#pragma unroll
        for (int j = 0; j < 2; ++j) {
            const bf16_t* p = Bb + (wn * 32 + j * 16 + lrow) * LDW + hf * 8;
            v8bf lo = *(const v8bf*)p;
            v8bf hi = *(const v8bf*)(p + 16);
            bfr[j] = __builtin_shufflevector(lo, hi, 0,1,2,3,4,5,6,7,8,9,10,11,12,13,14,15);
        }
#pragma unroll
        for (int i = 0; i < 2; ++i)
#pragma unroll
            for (int j = 0; j < 2; ++j)
                acc[i][j] = __builtin_amdgcn_wmma_f32_16x16x32_bf16(
                    false, afr[i], false, bfr[j], (short)0, acc[i][j], false, false);
    };

#if USE_ASYNC_LDS
    // -------- async global->LDS, double-buffered --------
    __shared__ bf16_t As[2][64 * LDW];
    __shared__ bf16_t Bs[2][128 * LDW];

    const bf16_t* gA  = A  + (size_t)(bm + aRow)  * K + seg;
    const bf16_t* gB0 = Bm + (size_t)(bn + bRow0) * K + seg;
    const bf16_t* gB1 = Bm + (size_t)(bn + bRow1) * K + seg;

    auto issue = [&](int kk) {
        const int buf = kk & 1;
        const int k0  = kk * KC;
        __builtin_amdgcn_global_load_async_to_lds_b128(
            (GAS v4i*)(gA + k0),  (LAS v4i*)&As[buf][aRow * LDW + seg], 0, 0);
        __builtin_amdgcn_global_load_async_to_lds_b128(
            (GAS v4i*)(gB0 + k0), (LAS v4i*)&Bs[buf][bRow0 * LDW + seg], 0, 0);
        __builtin_amdgcn_global_load_async_to_lds_b128(
            (GAS v4i*)(gB1 + k0), (LAS v4i*)&Bs[buf][bRow1 * LDW + seg], 0, 0);
    };

    issue(0);
    for (int kk = 0; kk < nk; ++kk) {
        if (kk + 1 < nk) {
            issue(kk + 1);                            // 3 more in flight -> <=6
            __builtin_amdgcn_s_wait_asynccnt(3);      // in-order: chunk kk landed
        } else {
            __builtin_amdgcn_s_wait_asynccnt(0);
        }
        __syncthreads();                              // all waves' chunk kk visible
        compute_chunk(As[kk & 1], Bs[kk & 1]);
        __syncthreads();                              // done reading before kk+2 overwrites
    }
#else
    // -------- fallback: reg-staged copy, single buffer --------
    __shared__ bf16_t As1[64 * LDW];
    __shared__ bf16_t Bs1[128 * LDW];

    auto loadA = [&](int k0) -> int4 {
        return *(const int4*)(A + (size_t)(bm + aRow) * K + k0 + seg);
    };
    auto loadB = [&](int row, int k0) -> int4 {
        return *(const int4*)(Bm + (size_t)(bn + row) * K + k0 + seg);
    };

    int4 aReg  = loadA(0);
    int4 bReg0 = loadB(bRow0, 0);
    int4 bReg1 = loadB(bRow1, 0);

    for (int kk = 0; kk < nk; ++kk) {
        *(int4*)(As1 + aRow  * LDW + seg) = aReg;
        *(int4*)(Bs1 + bRow0 * LDW + seg) = bReg0;
        *(int4*)(Bs1 + bRow1 * LDW + seg) = bReg1;

        if (kk + 1 < nk) {
            int k0 = (kk + 1) * KC;
            aReg  = loadA(k0);
            bReg0 = loadB(bRow0, k0);
            bReg1 = loadB(bRow1, k0);
        }
        if (kk + 2 < nk) {
            int kp = (kk + 2) * KC;
            __builtin_prefetch(Bm + (size_t)(bn + bRow0) * K + kp + seg, 0, 1);
            __builtin_prefetch(Bm + (size_t)(bn + bRow1) * K + kp + seg, 0, 1);
        }
        __syncthreads();
        compute_chunk(As1, Bs1);
        __syncthreads();
    }
#endif

    // ------------------------- epilogue -------------------------
    // C/D layout: element r -> m = (lane/16)*8 + r ; n = lane%16
#pragma unroll
    for (int i = 0; i < 2; ++i) {
#pragma unroll
        for (int j = 0; j < 2; ++j) {
            const int gm0 = bm + wm * 32 + i * 16 + hf * 8;
            const int gn  = bn + wn * 32 + j * 16 + lrow;
            if (MODE == 1) {
                const float w0 = wv2h[gn * 2 + 0];
                const float w1 = wv2h[gn * 2 + 1];
#pragma unroll
                for (int r = 0; r < 8; ++r) {
                    int gm = gm0 + r;
                    if (gm < M) {
                        const float* vp = vel + ((size_t)t * BATCH + gm) * 2;
                        float v = acc[i][j][r] + vp[0] * w0 + vp[1] * w1;
                        v = v > 0.f ? v : 0.f;
                        hout[(size_t)gm * NGG + gn] = (bf16_t)v;
                    }
                }
            } else if (MODE == 0) {
#pragma unroll
                for (int r = 0; r < 8; ++r) {
                    int gm = gm0 + r;
                    if (gm < M) hout[(size_t)gm * NGG + gn] = (bf16_t)acc[i][j][r];
                }
            } else {
#pragma unroll
                for (int r = 0; r < 8; ++r) {
                    int gm = gm0 + r;
                    if (gm < M) {
                        float v = acc[i][j][r];
                        if (gn < NPP)
                            out_pl[((size_t)t * BATCH + gm) * NPP + gn] = v;
                        else
                            out_im[((size_t)t * BATCH + gm) * IMGD + (gn - NPP)] =
                                1.0f / (1.0f + __expf(-v));
                    }
                }
            }
        }
    }
}

// ------------------------- launcher -------------------------

extern "C" void kernel_launch(void* const* d_in, const int* in_sizes, int n_in,
                              void* d_out, int out_size, void* d_ws, size_t ws_size,
                              hipStream_t stream) {
    (void)in_sizes; (void)n_in; (void)out_size; (void)ws_size;
    const float* image     = (const float*)d_in[0];   // [T,B,512]
    const float* vel       = (const float*)d_in[1];   // [T,B,2]
    const float* init_actv = (const float*)d_in[2];   // [B,512]
    const float* W_enc_pc  = (const float*)d_in[3];   // [4096,512]
    const float* W_enc_img = (const float*)d_in[4];   // [4096,512]
    const float* W_v2h     = (const float*)d_in[5];   // [4096,2]
    const float* W_h2h     = (const float*)d_in[6];   // [4096,4096]
    const float* W_dec_pc  = (const float*)d_in[7];   // [512,4096]
    const float* W_dec_img = (const float*)d_in[8];   // [512,4096]
    float* out = (float*)d_out;

    // workspace layout (bf16); A-side buffers sized for 256 rows
    char* ws = (char*)d_ws;
    size_t off = 0;
    bf16_t* Whh  = (bf16_t*)(ws + off); off += (size_t)NGG * NGG * 2;    // 32 MB
    bf16_t* Wenc = (bf16_t*)(ws + off); off += (size_t)NGG * 1024 * 2;   // 8 MB
    bf16_t* Wdec = (bf16_t*)(ws + off); off += (size_t)1024 * NGG * 2;   // 8 MB
    bf16_t* A0   = (bf16_t*)(ws + off); off += (size_t)256 * 1024 * 2;   // 512 KB
    bf16_t* h0b  = (bf16_t*)(ws + off); off += (size_t)256 * NGG * 2;    // 2 MB
    bf16_t* h1b  = (bf16_t*)(ws + off); off += (size_t)256 * NGG * 2;    // 2 MB

    const int TH = 256;
    cvt_f32_bf16<<<(NGG * NGG + TH - 1) / TH, TH, 0, stream>>>(W_h2h, Whh, NGG * NGG);
    pack_enc_kernel<<<(NGG * 1024 + TH - 1) / TH, TH, 0, stream>>>(W_enc_pc, W_enc_img, Wenc);
    cvt_f32_bf16<<<(NPP * NGG + TH - 1) / TH, TH, 0, stream>>>(W_dec_pc, Wdec, NPP * NGG);
    cvt_f32_bf16<<<(IMGD * NGG + TH - 1) / TH, TH, 0, stream>>>(W_dec_img, Wdec + (size_t)NPP * NGG, IMGD * NGG);
    pack_a0_kernel<<<(BATCH * 1024 + TH - 1) / TH, TH, 0, stream>>>(init_actv, image, A0);

    const dim3 gridStep(NGG / 128, 4);    // 32 x 4 (covers 256 M rows, stores guarded to 200)
    const dim3 gridDec(1024 / 128, 4);    //  8 x 4

    gemm_wmma<0><<<gridStep, TH, 0, stream>>>(A0, Wenc, BATCH, 1024, 0,
                                              nullptr, nullptr, h0b, nullptr, nullptr);

    float* out_place = out;
    float* out_image = out + (size_t)T_STEPS * BATCH * NPP;

    for (int t = 0; t < T_STEPS; ++t) {
        const bf16_t* hin = (t & 1) ? h1b : h0b;
        bf16_t*       hnw = (t & 1) ? h0b : h1b;
        gemm_wmma<1><<<gridStep, TH, 0, stream>>>(hin, Whh, BATCH, NGG, t,
                                                  vel, W_v2h, hnw, nullptr, nullptr);
        gemm_wmma<2><<<gridDec, TH, 0, stream>>>(hnw, Wdec, BATCH, NGG, t,
                                                 nullptr, nullptr, nullptr, out_place, out_image);
    }
}
